// MyRNN_12635793785049
// MI455X (gfx1250) — matrix operand order repaired and checked by reading, not compiled, not run on hardware
//
#include <hip/hip_runtime.h>
#include <hip/hip_bf16.h>
#include <math.h>

// ---------------------------------------------------------------------------
// Problem constants (from reference): BATCH=32, SEQ=80, VOCAB=10000, EMB=50,
// U1_DIM = U2_DIM = 4096.
// ---------------------------------------------------------------------------
#define BATCH 32
#define SEQ   80
#define EMB   50
#define DIM   4096

typedef __attribute__((ext_vector_type(16))) __bf16 v16bf;
typedef __attribute__((ext_vector_type(8)))  float  v8f;

union AFrag { v16bf v; uint4 q[2]; };
union BFrag { v16bf v; uint4 q[2]; };

// ---------------------------------------------------------------------------
// Kernel 1: fp32 [K=4096][N=4096] row-major  ->  bf16 transposed [N][K].
// Transposing lets each WMMA B-fragment lane read two runs of 8 contiguous bf16 (16B each).
// Tiled LDS transpose, coalesced on both sides.
// ---------------------------------------------------------------------------
__global__ __launch_bounds__(256)
void k_convT(const float* __restrict__ src, __bf16* __restrict__ dst) {
    __shared__ float tile[32][33];
    const int tx = threadIdx.x;           // 0..31
    const int ty = threadIdx.y;           // 0..7
    const int n0 = blockIdx.x * 32;
    const int k0 = blockIdx.y * 32;
    #pragma unroll
    for (int i = 0; i < 32; i += 8)
        tile[ty + i][tx] = src[(size_t)(k0 + ty + i) * DIM + (n0 + tx)];
    __syncthreads();
    #pragma unroll
    for (int i = 0; i < 32; i += 8)
        dst[(size_t)(n0 + ty + i) * DIM + (k0 + tx)] = (__bf16)tile[tx][ty + i];
}

// ---------------------------------------------------------------------------
// Kernel 2: zero the bf16 hidden-state buffers (d_ws is poisoned 0xAA).
// ---------------------------------------------------------------------------
__global__ void k_zero_bf16(__bf16* p, int n) {
    int i = blockIdx.x * blockDim.x + threadIdx.x;
    if (i < n) p[i] = (__bf16)0.0f;
}

// ---------------------------------------------------------------------------
// Kernel 3: XW1[t][b][n] = b1[n] + sum_e emb[tok(b,t)][e] * W1[e][n]
// Non-recurrent term for all time steps; fp32 precompute (K=50 only).
// Grid: (DIM/256, SEQ), block 256. Each block handles all 32 batch rows so
// each W1 value is reused 32x from registers.
// ---------------------------------------------------------------------------
__global__ __launch_bounds__(256)
void k_xw1(const int* __restrict__ tokens, const float* __restrict__ emb,
           const float* __restrict__ W1, const float* __restrict__ b1,
           float* __restrict__ xw1) {
    const int t = blockIdx.y;
    const int n = blockIdx.x * 256 + threadIdx.x;
    __shared__ float xs[BATCH][EMB];
    for (int i = threadIdx.x; i < BATCH * EMB; i += 256) {
        int b = i / EMB, e = i % EMB;
        int tok = tokens[b * SEQ + t];
        xs[b][e] = emb[(size_t)tok * EMB + e];
    }
    __syncthreads();
    float acc[BATCH];
    const float bias = b1[n];
    #pragma unroll
    for (int b = 0; b < BATCH; ++b) acc[b] = bias;
    for (int e = 0; e < EMB; ++e) {
        const float w = W1[(size_t)e * DIM + n];
        #pragma unroll
        for (int b = 0; b < BATCH; ++b) acc[b] = fmaf(xs[b][e], w, acc[b]);
    }
    #pragma unroll
    for (int b = 0; b < BATCH; ++b)
        xw1[((size_t)t * BATCH + b) * DIM + n] = acc[b];
}

// ---------------------------------------------------------------------------
// WMMA fragment loaders (CDNA5 16-bit layouts, ISA 7.12.2):
//   A (16x32, M x K): lane l -> row m = l&15; elements e: two contiguous
//     8-element runs: K = k0 + 8*half + (e&7) for e<8, +16 more for e>=8.
//   B (32x16, K x N), stored transposed BT[N][K]: lane l -> col n = l&15;
//     elements e: same K order as A (two 8-element runs of 16 bytes).
// ---------------------------------------------------------------------------
__device__ __forceinline__ v16bf load_a_frag(const __bf16* __restrict__ A,
                                             int m, int k0, int half) {
    AFrag a;
    const __bf16* p = A + (size_t)m * DIM + k0 + 8 * half;
    a.q[0] = *(const uint4*)(p);        // K = k0+8h .. k0+8h+7
    a.q[1] = *(const uint4*)(p + 16);   // K = k0+16+8h .. k0+16+8h+7
    return a.v;
}

__device__ __forceinline__ v16bf load_b_frag(const __bf16* __restrict__ BT,
                                             int n, int k0, int half) {
    BFrag b;
    const __bf16* p = BT + (size_t)n * DIM + k0 + 8 * half;
    b.q[0] = *(const uint4*)(p);        // K = k0+8h .. k0+8h+7
    b.q[1] = *(const uint4*)(p + 16);   // K = k0+16+8h .. k0+16+8h+7
    return b.v;
}

// ---------------------------------------------------------------------------
// Kernel 4: one RNN layer step.
//   out[32][4096] = tanh( (pre? pre : 0) + (bias? bias : 0)
//                         + A1[32][4096] @ B1  (+ A2 @ B2) )
// B matrices are pre-transposed bf16 (BT[N][K]); A and out are bf16 states.
// Wave covers M=32 (2 tiles) x N=32 (2 tiles) -> 4 f32 accumulators.
// Block = 4 waves = 128 threads = 128 columns; grid.x = DIM/128 = 32.
// All weight streams hit the 192MB L2 (bf16 weights total ~100MB).
// ---------------------------------------------------------------------------
__global__ __launch_bounds__(128)
void k_step(const __bf16* __restrict__ A1, const __bf16* __restrict__ B1T,
            const __bf16* __restrict__ A2, const __bf16* __restrict__ B2T,
            const float* __restrict__ pre, const float* __restrict__ bias,
            __bf16* __restrict__ out) {
    const int wave = threadIdx.x >> 5;
    const int lane = threadIdx.x & 31;
    const int half = lane >> 4;
    const int l16  = lane & 15;
    const int ncol0 = blockIdx.x * 128 + wave * 32;

    v8f acc[2][2];
    #pragma unroll
    for (int mt = 0; mt < 2; ++mt)
        #pragma unroll
        for (int nt = 0; nt < 2; ++nt)
            acc[mt][nt] = (v8f){0.f,0.f,0.f,0.f,0.f,0.f,0.f,0.f};

    // ---- GEMM 1: A1 @ B1 ----
    for (int k0 = 0; k0 < DIM; k0 += 32) {
        v16bf a0 = load_a_frag(A1, 0  + l16, k0, half);
        v16bf a1 = load_a_frag(A1, 16 + l16, k0, half);
        v16bf b0 = load_b_frag(B1T, ncol0 + l16,      k0, half);
        v16bf b1 = load_b_frag(B1T, ncol0 + 16 + l16, k0, half);
        acc[0][0] = __builtin_amdgcn_wmma_f32_16x16x32_bf16(false, a0, false, b0, (short)0, acc[0][0], false, false);
        acc[0][1] = __builtin_amdgcn_wmma_f32_16x16x32_bf16(false, a0, false, b1, (short)0, acc[0][1], false, false);
        acc[1][0] = __builtin_amdgcn_wmma_f32_16x16x32_bf16(false, a1, false, b0, (short)0, acc[1][0], false, false);
        acc[1][1] = __builtin_amdgcn_wmma_f32_16x16x32_bf16(false, a1, false, b1, (short)0, acc[1][1], false, false);
    }

    // ---- optional GEMM 2: A2 @ B2 (wave-uniform branch) ----
    if (A2 != nullptr) {
        for (int k0 = 0; k0 < DIM; k0 += 32) {
            v16bf a0 = load_a_frag(A2, 0  + l16, k0, half);
            v16bf a1 = load_a_frag(A2, 16 + l16, k0, half);
            v16bf b0 = load_b_frag(B2T, ncol0 + l16,      k0, half);
            v16bf b1 = load_b_frag(B2T, ncol0 + 16 + l16, k0, half);
            acc[0][0] = __builtin_amdgcn_wmma_f32_16x16x32_bf16(false, a0, false, b0, (short)0, acc[0][0], false, false);
            acc[0][1] = __builtin_amdgcn_wmma_f32_16x16x32_bf16(false, a0, false, b1, (short)0, acc[0][1], false, false);
            acc[1][0] = __builtin_amdgcn_wmma_f32_16x16x32_bf16(false, a1, false, b0, (short)0, acc[1][0], false, false);
            acc[1][1] = __builtin_amdgcn_wmma_f32_16x16x32_bf16(false, a1, false, b1, (short)0, acc[1][1], false, false);
        }
    }

    // ---- epilogue: C/D layout lane l: n = l&15; VGPR r: m = r + 8*half ----
    #pragma unroll
    for (int mt = 0; mt < 2; ++mt) {
        #pragma unroll
        for (int nt = 0; nt < 2; ++nt) {
            const int n = ncol0 + nt * 16 + l16;
            #pragma unroll
            for (int r = 0; r < 8; ++r) {
                const int m = mt * 16 + half * 8 + r;
                float v = acc[mt][nt][r];
                if (pre)  v += pre[(size_t)m * DIM + n];
                if (bias) v += bias[n];
                v = tanhf(v);
                out[(size_t)m * DIM + n] = (__bf16)v;
            }
        }
    }
}

// ---------------------------------------------------------------------------
// Kernel 5: out[b] = sigmoid( h2[b] . Wfc + bfc ). One block per batch row.
// ---------------------------------------------------------------------------
__global__ __launch_bounds__(256)
void k_final(const __bf16* __restrict__ h2, const float* __restrict__ Wfc,
             const float* __restrict__ bfc, float* __restrict__ out) {
    const int b = blockIdx.x;
    float s = 0.f;
    for (int n = threadIdx.x; n < DIM; n += 256)
        s += (float)h2[(size_t)b * DIM + n] * Wfc[n];
    __shared__ float red[256];
    red[threadIdx.x] = s;
    __syncthreads();
    for (int off = 128; off > 0; off >>= 1) {
        if (threadIdx.x < off) red[threadIdx.x] += red[threadIdx.x + off];
        __syncthreads();
    }
    if (threadIdx.x == 0) {
        float z = red[0] + bfc[0];
        out[b] = 1.f / (1.f + expf(-z));
    }
}

// ---------------------------------------------------------------------------
// Host launcher
// ---------------------------------------------------------------------------
extern "C" void kernel_launch(void* const* d_in, const int* in_sizes, int n_in,
                              void* d_out, int out_size, void* d_ws, size_t ws_size,
                              hipStream_t stream) {
    (void)in_sizes; (void)n_in; (void)out_size; (void)ws_size;
    const int*   tokens = (const int*)  d_in[0];   // [32][80]
    const float* emb    = (const float*)d_in[1];   // [10000][50]
    const float* W1     = (const float*)d_in[2];   // [50][4096]
    const float* U1     = (const float*)d_in[3];   // [4096][4096]
    const float* b1     = (const float*)d_in[4];   // [4096]
    const float* W2     = (const float*)d_in[5];   // [4096][4096]
    const float* U2     = (const float*)d_in[6];   // [4096][4096]
    const float* b2     = (const float*)d_in[7];   // [4096]
    const float* Wfc    = (const float*)d_in[8];   // [4096]
    const float* bfc    = (const float*)d_in[9];   // [1]
    float* out = (float*)d_out;                    // [32]

    // Workspace carve-up (256B-aligned chunks).
    char*  ws  = (char*)d_ws;
    size_t off = 0;
    auto carve = [&](size_t bytes) -> void* {
        void* p = ws + off;
        off += (bytes + 255) & ~(size_t)255;
        return p;
    };
    const size_t WBYTES = (size_t)DIM * DIM * sizeof(__bf16);      // 32 MiB each
    __bf16* U1bT = (__bf16*)carve(WBYTES);
    __bf16* W2bT = (__bf16*)carve(WBYTES);
    __bf16* U2bT = (__bf16*)carve(WBYTES);
    float*  XW1  = (float*) carve((size_t)SEQ * BATCH * DIM * sizeof(float)); // 40 MiB
    __bf16* h1[2], *h2[2];
    h1[0] = (__bf16*)carve((size_t)BATCH * DIM * sizeof(__bf16));
    h1[1] = (__bf16*)carve((size_t)BATCH * DIM * sizeof(__bf16));
    h2[0] = (__bf16*)carve((size_t)BATCH * DIM * sizeof(__bf16));
    h2[1] = (__bf16*)carve((size_t)BATCH * DIM * sizeof(__bf16));

    // 1) Convert + transpose the three big weight matrices to bf16 (fits L2).
    {
        dim3 g(DIM / 32, DIM / 32), blk(32, 8);
        k_convT<<<g, blk, 0, stream>>>(U1, U1bT);
        k_convT<<<g, blk, 0, stream>>>(W2, W2bT);
        k_convT<<<g, blk, 0, stream>>>(U2, U2bT);
    }

    // 2) Zero initial hidden states (d_ws is poisoned).
    {
        const int n = BATCH * DIM;
        k_zero_bf16<<<(n + 255) / 256, 256, 0, stream>>>(h1[0], n);
        k_zero_bf16<<<(n + 255) / 256, 256, 0, stream>>>(h2[0], n);
    }

    // 3) Precompute XW1[t] = emb[tokens]@W1 + b1 for all steps.
    {
        dim3 g(DIM / 256, SEQ);
        k_xw1<<<g, 256, 0, stream>>>(tokens, emb, W1, b1, XW1);
    }

    // 4) Sequential scan: 2 WMMA step-kernels per time step (double-buffered).
    int p = 0;
    for (int t = 0; t < SEQ; ++t) {
        const float* pre = XW1 + (size_t)t * BATCH * DIM;
        // h1' = tanh(xw1[t] + h1 @ U1)          (b1 already folded into pre)
        k_step<<<DIM / 128, 128, 0, stream>>>(h1[p], U1bT,
                                              nullptr, nullptr,
                                              pre, nullptr, h1[1 - p]);
        // h2' = tanh(h1' @ W2 + h2 @ U2 + b2)
        k_step<<<DIM / 128, 128, 0, stream>>>(h1[1 - p], W2bT,
                                              h2[p], U2bT,
                                              nullptr, b2, h2[1 - p]);
        p ^= 1;
    }

    // 5) Final sigmoid head.
    k_final<<<BATCH, 256, 0, stream>>>(h2[p], Wfc, bfc, out);
}
